// GCNNet_7859790152294
// MI455X (gfx1250) — compile-verified, matrix operand order checked by reading, not run on hardware
//
#include <hip/hip_runtime.h>
#include <hip/hip_bf16.h>
#include <math.h>

#define N_NODES     50000
#define N_EDGES     800000
#define NUM_FEATS   128
#define HIDDEN      256
#define NUM_CLASSES 40

typedef __attribute__((ext_vector_type(16))) _Float16 v16h;
typedef __attribute__((ext_vector_type(8)))  float    v8f;

// ---------------------------------------------------------------------------
// Edge-parallel scatter-add: agg[dst] += ew * x[src]. One thread = 1 edge x 4
// features (float4 gather + 4 f32 atomics). Memory/atomic bound by design.
// ---------------------------------------------------------------------------
__global__ void scatter_kernel(const float* __restrict__ x,
                               const int*   __restrict__ ei,
                               const float* __restrict__ ew,
                               float*       __restrict__ agg,
                               int D) {
    const int vecPerEdge = D >> 2;                 // float4 chunks per edge
    const int gid = blockIdx.x * blockDim.x + threadIdx.x;
    const int total = N_EDGES * vecPerEdge;        // <= 51.2M, fits int
    if (gid >= total) return;
    const int e  = gid / vecPerEdge;
    const int f4 = (gid - e * vecPerEdge) << 2;
    const int src = ei[e];
    const int dst = ei[N_EDGES + e];
    const float w = ew[e];
    const float4 v = *(const float4*)(x + (long long)src * D + f4);
    float* o = agg + (long long)dst * D + f4;
    atomicAdd(o + 0, v.x * w);
    atomicAdd(o + 1, v.y * w);
    atomicAdd(o + 2, v.z * w);
    atomicAdd(o + 3, v.w * w);
}

// ---------------------------------------------------------------------------
// WMMA GEMM: C[M=50000, Ncols] = A[M, K] * W[Ncols, K]^T  (row-major all).
// One wave per 16x16 output tile. f32 inputs converted to f16 fragments,
// f32 accumulation via v_wmma_f32_16x16x32_f16. Optional fused ReLU.
//
// A fragment (16x32 f16, per ISA 7.12.2): lane l holds row (l&15);
//   element e -> K = k0 + 16*(e>=8) + 8*(l>=16) + (e&7).
// B fragment (32x16): lane l holds column (l&15), same K mapping.
// C/D (16x16 f32): lane l, vgpr r -> row = 8*(l>=16)+r, col = l&15.
// ---------------------------------------------------------------------------
template <int K>
__global__ __launch_bounds__(32)
void gemm_wmma_kernel(const float* __restrict__ A,
                      const float* __restrict__ W,
                      float*       __restrict__ C,
                      int Ncols, int do_relu) {
    const int l     = threadIdx.x;        // 0..31 (wave32)
    const int rl    = l & 15;
    const int khalf = (l >> 4) * 8;
    const int row0  = blockIdx.x * 16;
    const int col0  = blockIdx.y * 16;
    const int col   = col0 + rl;

    v8f acc = {};
    const float* arow = A + (long long)(row0 + rl) * K + khalf;
    const float* wrow = W + (long long)col * K + khalf;  // guarded below

    #pragma unroll
    for (int k0 = 0; k0 < K; k0 += 32) {
        v16h af;
        #pragma unroll
        for (int e = 0; e < 16; ++e) {
            const int kk = k0 + ((e & 8) ? 16 : 0) + (e & 7);
            af[e] = (_Float16)arow[kk];
        }
        v16h bf = {};
        if (col < Ncols) {                 // pad columns beyond Ncols with 0
            #pragma unroll
            for (int e = 0; e < 16; ++e) {
                const int kk = k0 + ((e & 8) ? 16 : 0) + (e & 7);
                bf[e] = (_Float16)wrow[kk];
            }
        }
        // EXEC is all-1s again here (guard closed) -> WMMA legal.
        acc = __builtin_amdgcn_wmma_f32_16x16x32_f16(
            /*neg_a=*/false, af, /*neg_b=*/false, bf,
            /*c_mod=*/(short)0, acc, /*reuse_a=*/false, /*reuse_b=*/false);
    }

    if (col < Ncols) {
        #pragma unroll
        for (int r = 0; r < 8; ++r) {
            float v = acc[r];
            if (do_relu) v = v > 0.0f ? v : 0.0f;
            C[(long long)(row0 + khalf + r) * Ncols + col] = v;
        }
    }
}

// ---------------------------------------------------------------------------
// Row-wise log_softmax over NUM_CLASSES=40. One thread per node (24 MB total
// traffic -> negligible).
// ---------------------------------------------------------------------------
__global__ void logsoftmax_kernel(const float* __restrict__ H,
                                  float*       __restrict__ out) {
    const int n = blockIdx.x * blockDim.x + threadIdx.x;
    if (n >= N_NODES) return;
    const float* h = H + (long long)n * NUM_CLASSES;
    float mx = -INFINITY;
    #pragma unroll
    for (int c = 0; c < NUM_CLASSES; ++c) mx = fmaxf(mx, h[c]);
    float s = 0.0f;
    #pragma unroll
    for (int c = 0; c < NUM_CLASSES; ++c) s += __expf(h[c] - mx);
    const float lse = __logf(s) + mx;
    float* o = out + (long long)n * NUM_CLASSES;
    #pragma unroll
    for (int c = 0; c < NUM_CLASSES; ++c) o[c] = h[c] - lse;
}

// ---------------------------------------------------------------------------
// Orchestration. Workspace layout (double buffer, agg2 aliases agg1's slot):
//   bufA: 50000*256 f32  (agg1 uses first half; agg2 uses all of it)
//   bufB: 50000*256 f32  (h1 uses all; h2 uses first 40 cols worth)
// Total required ws: 102.4 MB.
// ---------------------------------------------------------------------------
extern "C" void kernel_launch(void* const* d_in, const int* in_sizes, int n_in,
                              void* d_out, int out_size, void* d_ws, size_t ws_size,
                              hipStream_t stream) {
    const float* x  = (const float*)d_in[0];
    const int*   ei = (const int*)  d_in[1];
    const float* ew = (const float*)d_in[2];
    const float* w1 = (const float*)d_in[3];   // [HIDDEN, NUM_FEATS]
    const float* w2 = (const float*)d_in[4];   // [NUM_CLASSES, HIDDEN]
    float* out = (float*)d_out;

    const size_t bufElems = (size_t)N_NODES * HIDDEN;
    float* bufA = (float*)d_ws;
    float* bufB = bufA + bufElems;
    float* agg1 = bufA;   // [N, 128]
    float* h1   = bufB;   // [N, 256]
    float* agg2 = bufA;   // [N, 256] (reuses agg1 slot after GEMM1)
    float* h2   = bufB;   // [N, 40]  (reuses h1 slot after scatter2)

    // ---- Layer 1: scatter-add over edges, then h1 = relu(agg1 @ w1^T) ----
    hipMemsetAsync(agg1, 0, (size_t)N_NODES * NUM_FEATS * sizeof(float), stream);
    {
        const int total = N_EDGES * (NUM_FEATS / 4);
        scatter_kernel<<<(total + 255) / 256, 256, 0, stream>>>(x, ei, ew, agg1, NUM_FEATS);
    }
    gemm_wmma_kernel<NUM_FEATS><<<dim3(N_NODES / 16, HIDDEN / 16), 32, 0, stream>>>(
        agg1, w1, h1, HIDDEN, /*do_relu=*/1);

    // ---- Layer 2: scatter-add over edges, then h2 = agg2 @ w2^T ----------
    hipMemsetAsync(agg2, 0, (size_t)N_NODES * HIDDEN * sizeof(float), stream);
    {
        const int total = N_EDGES * (HIDDEN / 4);
        scatter_kernel<<<(total + 255) / 256, 256, 0, stream>>>(h1, ei, ew, agg2, HIDDEN);
    }
    gemm_wmma_kernel<HIDDEN><<<dim3(N_NODES / 16, 3 /* 48 cols padded */), 32, 0, stream>>>(
        agg2, w2, h2, NUM_CLASSES, /*do_relu=*/0);

    // ---- log_softmax --------------------------------------------------------
    logsoftmax_kernel<<<(N_NODES + 255) / 256, 256, 0, stream>>>(h2, out);
}
